// ProbAttention_12764642804171
// MI455X (gfx1250) — matrix operand order repaired and checked by reading, not compiled, not run on hardware
//
#include <hip/hip_runtime.h>
#include <hip/hip_bf16.h>
#include <math.h>

typedef __attribute__((ext_vector_type(16))) _Float16 v16h;
typedef __attribute__((ext_vector_type(8)))  float    v8f;

#define WS   9
#define LSEQ 81      // sequence length (= WS*WS)
#define DIM  64      // head dim
#define NH   8       // heads
#define NS   41      // n_samp (even keys)
#define UTOP 10      // top-k queries (u)
#define LPAD 96      // padded rows (6 tiles of 16)
#define SPAD 48      // padded even-key count (3 tiles of 16)

// ---- LDS layout (bytes) ----
#define QH_OFF    0                        // f16 [96][64]
#define KH_OFF    12288                    // f16 [96][64]
#define VH_OFF    24576                    // f16 [96][64]
#define SCR_OFF   36864
// phase 1-3 view of scratch
#define QKS_OFF   SCR_OFF                  // f32 [96][48]
#define M_OFF     (SCR_OFF + 18432)        // f32 [96]
// phase 4-6 view of scratch (same bytes, later lifetime)
#define QRED_OFF  SCR_OFF                  // f16 [16][64]
#define SCO_OFF   (SCR_OFF + 2048)         // f32 [16][96]
#define ATTNH_OFF (SCR_OFF + 2048 + 6144)  // f16 [16][96]
#define UPD_OFF   (SCR_OFF + 2048 + 6144 + 3072) // f32 [16][64]
// persistent small buffers (beyond both scratch views)
#define PERS_OFF  (SCR_OFF + 18816)
#define MTOP_OFF  PERS_OFF                 // int [16]
#define RANK_OFF  (PERS_OFF + 64)          // int [96]
#define SMEM_BYTES (RANK_OFF + 384)        // 56128 B

__global__ __launch_bounds__(256) void prob_attn_kernel(
    const float* __restrict__ qg_, const float* __restrict__ kg_,
    const float* __restrict__ vg_, const float* __restrict__ bias,
    float* __restrict__ out_ctx, float* __restrict__ out_attn)
{
    __shared__ __align__(16) unsigned char smem[SMEM_BYTES];
    _Float16* Qh   = (_Float16*)(smem + QH_OFF);
    _Float16* Kh   = (_Float16*)(smem + KH_OFF);
    _Float16* Vh   = (_Float16*)(smem + VH_OFF);
    float*    QKS  = (float*)(smem + QKS_OFF);
    float*    Mb   = (float*)(smem + M_OFF);
    _Float16* Qred = (_Float16*)(smem + QRED_OFF);
    float*    Sco  = (float*)(smem + SCO_OFF);
    _Float16* Ath  = (_Float16*)(smem + ATTNH_OFF);
    float*    Upd  = (float*)(smem + UPD_OFF);
    int*      Mtop = (int*)(smem + MTOP_OFF);
    int*      Rank = (int*)(smem + RANK_OFF);

    const int tid  = threadIdx.x;
    const int wave = tid >> 5;
    const int lane = tid & 31;
    const int nl   = lane & 15;   // row/col within half-wave per WMMA layout
    const int hi   = lane >> 4;   // half-wave select

    const int b = blockIdx.x >> 3;
    const int h = blockIdx.x & 7;
    const size_t gbase = (size_t)b * (LSEQ * NH * DIM) + (size_t)h * DIM;
    const float* Qg = qg_ + gbase;
    const float* Kg = kg_ + gbase;
    const float* Vg = vg_ + gbase;

    // zero all f16 tile buffers (padding rows must be 0 for WMMA)
    for (int i = tid; i < 36864 / 4; i += 256) ((unsigned int*)smem)[i] = 0u;
    __syncthreads();

    // load fp32 -> f16 LDS tiles (row-major, 64 dims contiguous)
    for (int idx = tid; idx < LSEQ * DIM; idx += 256) {
        int l = idx >> 6, d = idx & 63;
        size_t gi = (size_t)l * (NH * DIM) + d;
        Qh[l * DIM + d] = (_Float16)Qg[gi];
        Kh[l * DIM + d] = (_Float16)Kg[gi];
        Vh[l * DIM + d] = (_Float16)Vg[gi];
    }
    __syncthreads();

    // ---- Phase 1: QK_s = Q @ K_even^T  (96x48 padded), v_wmma f16->f32 ----
    for (int t = wave; t < 18; t += 8) {
        int mt = t / 3, st = t % 3;
        v8f c = {};
        #pragma unroll
        for (int k0 = 0; k0 < DIM; k0 += 32) {
            v16h a, bm;
            #pragma unroll
            for (int j = 0; j < 16; ++j) {
                int kk = (j & 7) + ((j >> 3) << 4) + hi * 8;   // ISA A-layout
                a[j] = Qh[(mt * 16 + nl) * DIM + k0 + kk];
            }
            int krow = 2 * (st * 16 + nl);                      // even-key gather (<=94, padded rows are 0)
            #pragma unroll
            for (int j = 0; j < 16; ++j)
                bm[j] = Kh[krow * DIM + k0 + (j & 7) + ((j >> 3) << 4) + hi * 8];      // ISA B-layout
            c = __builtin_amdgcn_wmma_f32_16x16x32_f16(false, a, false, bm,
                                                       (short)0, c, false, false);
        }
        #pragma unroll
        for (int r = 0; r < 8; ++r)
            QKS[(mt * 16 + r + hi * 8) * SPAD + st * 16 + nl] = c[r];
    }
    __syncthreads();

    // ---- Phase 2: sparsity metric M = max_s - sum_s / L ----
    if (tid < LSEQ) {
        float mx = -INFINITY, sm = 0.f;
        for (int s = 0; s < NS; ++s) {
            float x = QKS[tid * SPAD + s];
            mx = fmaxf(mx, x); sm += x;
        }
        Mb[tid] = mx - sm * (1.0f / (float)LSEQ);
    }
    __syncthreads();

    // ---- Phase 3: top-U selection (stable, lowest index on ties) ----
    if (tid == 0) {
        for (int i = 0; i < UTOP; ++i) {
            float best = -INFINITY; int bi = 0;
            for (int l = 0; l < LSEQ; ++l)
                if (Mb[l] > best) { best = Mb[l]; bi = l; }
            Mtop[i] = bi; Mb[bi] = -INFINITY;
        }
        for (int l = 0; l < LPAD; ++l) Rank[l] = -1;
        for (int i = 0; i < UTOP; ++i) Rank[Mtop[i]] = i;
    }
    __syncthreads();

    // ---- Phase 4a: zero Q_red + attn-f16, gather Q_red ----
    for (int i = tid; i < 2048 / 4; i += 256) ((unsigned int*)(smem + QRED_OFF))[i]  = 0u;
    for (int i = tid; i < 3072 / 4; i += 256) ((unsigned int*)(smem + ATTNH_OFF))[i] = 0u;
    __syncthreads();
    for (int idx = tid; idx < UTOP * DIM; idx += 256) {
        int i = idx >> 6, d = idx & 63;
        Qred[i * DIM + d] = Qh[Mtop[i] * DIM + d];
    }
    __syncthreads();

    // ---- Phase 4b: scores = Q_red @ K^T  (16 x 96) ----
    if (wave < 6) {
        int nt = wave;
        v8f c = {};
        #pragma unroll
        for (int k0 = 0; k0 < DIM; k0 += 32) {
            v16h a, bm;
            #pragma unroll
            for (int j = 0; j < 16; ++j) {
                int kk = (j & 7) + ((j >> 3) << 4) + hi * 8;
                a[j] = Qred[nl * DIM + k0 + kk];
            }
            #pragma unroll
            for (int j = 0; j < 16; ++j)
                bm[j] = Kh[(nt * 16 + nl) * DIM + k0 + (j & 7) + ((j >> 3) << 4) + hi * 8];
            c = __builtin_amdgcn_wmma_f32_16x16x32_f16(false, a, false, bm,
                                                       (short)0, c, false, false);
        }
        #pragma unroll
        for (int r = 0; r < 8; ++r)
            Sco[(r + hi * 8) * LPAD + nt * 16 + nl] = c[r];
    }
    __syncthreads();

    // ---- Phase 4c: +bias, *scale, softmax; emit attn (global + f16 LDS) ----
    if (tid < UTOP) {
        int i = tid;
        int qi = i / WS, qj = i % WS;
        float mx = -INFINITY;
        for (int kk = 0; kk < LSEQ; ++kk) {
            int ki = kk / WS, kj = kk % WS;
            int bidx = (qi - ki + WS - 1) * (2 * WS - 1) + (qj - kj + WS - 1);
            float s = (Sco[i * LPAD + kk] + bias[bidx]) * 0.125f;   // 1/sqrt(64)
            Sco[i * LPAD + kk] = s;
            mx = fmaxf(mx, s);
        }
        float sum = 0.f;
        for (int kk = 0; kk < LSEQ; ++kk) {
            float e = __expf(Sco[i * LPAD + kk] - mx);
            Sco[i * LPAD + kk] = e; sum += e;
        }
        float inv = 1.0f / sum;
        float* ao = out_attn + ((size_t)(b * NH + h) * UTOP + i) * LSEQ;
        for (int kk = 0; kk < LSEQ; ++kk) {
            float a = Sco[i * LPAD + kk] * inv;
            Ath[i * LPAD + kk] = (_Float16)a;
            ao[kk] = a;
        }
    }
    __syncthreads();

    // ---- Phase 5: upd = attn @ V  (16 x 64, K=96 padded) ----
    if (wave < 4) {
        int nt = wave;
        v8f c = {};
        #pragma unroll
        for (int k0 = 0; k0 < LPAD; k0 += 32) {
            v16h a, bm;
            #pragma unroll
            for (int j = 0; j < 16; ++j) {
                int kk = (j & 7) + ((j >> 3) << 4) + hi * 8;
                a[j] = Ath[nl * LPAD + k0 + kk];
            }
            #pragma unroll
            for (int j = 0; j < 16; ++j)
                bm[j] = Vh[(k0 + (j & 7) + ((j >> 3) << 4) + hi * 8) * DIM + nt * 16 + nl];
            c = __builtin_amdgcn_wmma_f32_16x16x32_f16(false, a, false, bm,
                                                       (short)0, c, false, false);
        }
        #pragma unroll
        for (int r = 0; r < 8; ++r)
            Upd[(r + hi * 8) * DIM + nt * 16 + nl] = c[r];
    }
    __syncthreads();

    // ---- Phase 6: fp32 cumsum of V + scatter-replace top-u rows ----
    if (tid < DIM) {
        int d = tid;
        float acc = 0.f;
        float* co = out_ctx + gbase;
        for (int l = 0; l < LSEQ; ++l) {
            acc += Vg[(size_t)l * (NH * DIM) + d];
            int r = Rank[l];
            co[(size_t)l * (NH * DIM) + d] = (r >= 0) ? Upd[r * DIM + d] : acc;
        }
    }
}

extern "C" void kernel_launch(void* const* d_in, const int* in_sizes, int n_in,
                              void* d_out, int out_size, void* d_ws, size_t ws_size,
                              hipStream_t stream) {
    const float* q    = (const float*)d_in[0];
    const float* k    = (const float*)d_in[1];
    const float* v    = (const float*)d_in[2];
    const float* bias = (const float*)d_in[3];
    const int B = in_sizes[0] / (LSEQ * NH * DIM);   // 1024
    float* out_ctx  = (float*)d_out;
    float* out_attn = out_ctx + (size_t)B * LSEQ * NH * DIM;
    prob_attn_kernel<<<dim3(B * NH), dim3(256), 0, stream>>>(q, k, v, bias,
                                                             out_ctx, out_attn);
}
